// MultiheadAttentionWithBias_42649025250079
// MI455X (gfx1250) — matrix operand order repaired and checked by reading, not compiled, not run on hardware
//
#include <hip/hip_runtime.h>
#include <hip/hip_bf16.h>

typedef __attribute__((ext_vector_type(16))) _Float16 v16h;
typedef __attribute__((ext_vector_type(8)))  _Float16 v8h;
typedef __attribute__((ext_vector_type(4)))  _Float16 v4h;
typedef __attribute__((ext_vector_type(8)))  float    v8f;
typedef __attribute__((ext_vector_type(4)))  float    f32x4;

constexpr int Bc  = 2;
constexpr int Lc  = 2048;
constexpr int Dc  = 1024;
constexpr int Hc  = 16;
constexpr int HDc = 64;

// ---------------------------------------------------------------------------
// Kernel 0: one-shot f32 -> f16 conversion (x and the four weight matrices).
// Everything stays L2-resident afterwards; removes all cvt work from the
// GEMM inner loops.
// ---------------------------------------------------------------------------
__global__ __launch_bounds__(256) void cvt_f16_kernel(
    const float* __restrict__ src, _Float16* __restrict__ dst, int n4)
{
    const int i = (blockIdx.x * 256 + threadIdx.x);
    if (i < n4) {
        f32x4 v = *(const f32x4*)(src + (size_t)i * 4);
        v4h o;
        o[0] = (_Float16)v[0]; o[1] = (_Float16)v[1];
        o[2] = (_Float16)v[2]; o[3] = (_Float16)v[3];
        *(v4h*)(dst + (size_t)i * 4) = o;
    }
}

// ---------------------------------------------------------------------------
// Kernel 1: QKV projections.  C = x @ W^T + b, all-f16 operands.
//   q,k stored [B,H,L,HD];  v stored transposed [B,H,HD,L].
// One wave computes a 32(M) x 64(N) tile: 8 WMMAs per K=32 step from
// 12 global_load_b128s.
// ---------------------------------------------------------------------------
__global__ __launch_bounds__(256) void qkv_proj_kernel(
    const _Float16* __restrict__ xh,
    const _Float16* __restrict__ Wqh, const float* __restrict__ bq,
    const _Float16* __restrict__ Wkh, const float* __restrict__ bk,
    const _Float16* __restrict__ Wvh, const float* __restrict__ bv,
    _Float16* __restrict__ qbuf, _Float16* __restrict__ kbuf,
    _Float16* __restrict__ vT)
{
    const int lane = threadIdx.x & 31;
    const int wid  = threadIdx.x >> 5;
    const int wg   = blockIdx.x * 8 + wid;       // 0 .. 6143
    const int proj = wg >> 11;                   // 0=q 1=k 2=v
    const int rem  = wg & 2047;
    const int m0   = (rem >> 4) * 32;            // 128 M-tiles of 32
    const int n0   = (rem & 15) * 64;

    const _Float16* W  = (proj == 0) ? Wqh : (proj == 1) ? Wkh : Wvh;
    const float*    bb = (proj == 0) ? bq  : (proj == 1) ? bk  : bv;

    const int colN = lane & 15;
    const int half = lane >> 4;
    const int roff = half * 8;
    const int kbA  = half * 8;                   // A-operand K base
    const int kbB  = half * 8;                   // B-operand K base

    const _Float16* xr0 = xh + (size_t)(m0 + colN) * Dc;
    const _Float16* xr1 = xh + (size_t)(m0 + 16 + colN) * Dc;

    v8f acc[8];
    #pragma unroll
    for (int i = 0; i < 8; ++i) acc[i] = {};

    for (int k0 = 0; k0 < Dc; k0 += 32) {
        v16h a0, a1;
        {
            v8h lo = *(const v8h*)(xr0 + k0 + kbA);
            v8h hi = *(const v8h*)(xr0 + k0 + kbA + 16);
            #pragma unroll
            for (int e = 0; e < 8; ++e) { a0[e] = lo[e]; a0[8 + e] = hi[e]; }
            lo = *(const v8h*)(xr1 + k0 + kbA);
            hi = *(const v8h*)(xr1 + k0 + kbA + 16);
            #pragma unroll
            for (int e = 0; e < 8; ++e) { a1[e] = lo[e]; a1[8 + e] = hi[e]; }
        }
        #pragma unroll
        for (int t = 0; t < 4; ++t) {
            const _Float16* wr = W + (size_t)(n0 + t * 16 + colN) * Dc + k0 + kbB;
            v16h bm;
            v8h lo = *(const v8h*)(wr);
            v8h hi = *(const v8h*)(wr + 16);
            #pragma unroll
            for (int e = 0; e < 8; ++e) { bm[e] = lo[e]; bm[8 + e] = hi[e]; }
            acc[t] = __builtin_amdgcn_wmma_f32_16x16x32_f16(
                false, a0, false, bm, (short)0, acc[t], false, false);
            acc[4 + t] = __builtin_amdgcn_wmma_f32_16x16x32_f16(
                false, a1, false, bm, (short)0, acc[4 + t], false, false);
        }
    }

    // ---- epilogue: + bias, store f16 with per-projection routing ----
    #pragma unroll
    for (int mi = 0; mi < 2; ++mi) {
        #pragma unroll
        for (int t = 0; t < 4; ++t) {
            const v8f a = acc[mi * 4 + t];
            const int n     = n0 + t * 16 + colN;
            const float bv_ = bb[n];
            const int h  = n >> 6;
            const int hd = n & 63;
            #pragma unroll
            for (int r = 0; r < 8; ++r) {
                const int m    = m0 + mi * 16 + roff + r;
                const int bidx = m >> 11;
                const int li   = m & (Lc - 1);
                const _Float16 val = (_Float16)(a[r] + bv_);
                if (proj == 2)
                    vT[(((size_t)bidx * Hc + h) * HDc + hd) * Lc + li] = val;
                else {
                    _Float16* dst = (proj == 0) ? qbuf : kbuf;
                    dst[(((size_t)bidx * Hc + h) * Lc + li) * HDc + hd] = val;
                }
            }
        }
    }
}

// ---------------------------------------------------------------------------
// Kernel 2: flash attention with additive rel_bias.
// One wave owns a 16-query tile of one (b,h); streams keys in chunks of 32.
// Per chunk: 4 WMMA (S = q k^T, K=64), online softmax, P->LDS->A, 4 WMMA (P v).
// ---------------------------------------------------------------------------
__global__ __launch_bounds__(256) void attn_kernel(
    const _Float16* __restrict__ qbuf, const _Float16* __restrict__ kbuf,
    const _Float16* __restrict__ vT,   const float* __restrict__ rel,
    _Float16* __restrict__ obuf)
{
    __shared__ alignas(16) _Float16 Plds[8][16][32];   // per-wave P staging

    const int lane = threadIdx.x & 31;
    const int wid  = threadIdx.x >> 5;
    const int wg   = blockIdx.x * 8 + wid;       // 0 .. 4095
    const int bh   = wg >> 7;                    // 0 .. 31
    const int q0   = (wg & 127) * 16;
    const int b    = bh >> 4;
    const int h    = bh & 15;

    const int colN = lane & 15;
    const int half = lane >> 4;
    const int roff = half * 8;
    const int kbA  = half * 8;
    const int kbB  = half * 8;
    const float scale = 0.125f;                  // 1/sqrt(64)

    // ---- Q as two A-operands (K = hd 0..31 and 32..63) ----
    const _Float16* qrow = qbuf + ((size_t)bh * Lc + q0 + colN) * HDc;
    v16h aq0, aq1;
    {
        v8h lo = *(const v8h*)(qrow + kbA);
        v8h hi = *(const v8h*)(qrow + kbA + 16);
        #pragma unroll
        for (int e = 0; e < 8; ++e) { aq0[e] = lo[e]; aq0[8 + e] = hi[e]; }
        lo = *(const v8h*)(qrow + 32 + kbA);
        hi = *(const v8h*)(qrow + 32 + kbA + 16);
        #pragma unroll
        for (int e = 0; e < 8; ++e) { aq1[e] = lo[e]; aq1[8 + e] = hi[e]; }
    }

    const float*    brow  = rel + ((size_t)h * Lc + (q0 + roff)) * Lc;
    const _Float16* kbase = kbuf + (size_t)bh * Lc * HDc;
    const _Float16* vbase = vT   + (size_t)bh * HDc * Lc;

    v8f m_run, l_run, o0 = {}, o1 = {}, o2 = {}, o3 = {};
    #pragma unroll
    for (int r = 0; r < 8; ++r) { m_run[r] = -1e30f; l_run[r] = 0.f; }

    for (int kk = 0; kk < Lc; kk += 32) {
        // ---- S = (q k^T) * scale + bias, as two 16x16 subtiles ----
        v8f st0 = {}, st1 = {};
        #pragma unroll
        for (int t = 0; t < 2; ++t) {
            const _Float16* krow = kbase + (size_t)(kk + t * 16 + colN) * HDc + kbB;
            v16h bm;
            v8h lo = *(const v8h*)(krow);
            v8h hi = *(const v8h*)(krow + 16);
            #pragma unroll
            for (int e = 0; e < 8; ++e) { bm[e] = lo[e]; bm[8 + e] = hi[e]; }
            v8f s = {};
            s = __builtin_amdgcn_wmma_f32_16x16x32_f16(
                false, aq0, false, bm, (short)0, s, false, false);
            lo = *(const v8h*)(krow + 32);
            hi = *(const v8h*)(krow + 48);
            #pragma unroll
            for (int e = 0; e < 8; ++e) { bm[e] = lo[e]; bm[8 + e] = hi[e]; }
            s = __builtin_amdgcn_wmma_f32_16x16x32_f16(
                false, aq1, false, bm, (short)0, s, false, false);

            const float* bp = brow + kk + t * 16 + colN;
            #pragma unroll
            for (int r = 0; r < 8; ++r)
                s[r] = s[r] * scale + bp[(size_t)r * Lc];
            if (t == 0) st0 = s; else st1 = s;
        }
        if (kk + 64 < Lc)
            __builtin_prefetch(brow + kk + 64 + colN, 0, 0);   // stream rel_bias

        // ---- online softmax (rows live in VGPR r across the 16-lane half) ----
        v8f cm;
        #pragma unroll
        for (int r = 0; r < 8; ++r) cm[r] = fmaxf(st0[r], st1[r]);
        #pragma unroll
        for (int mask = 1; mask <= 8; mask <<= 1)
            #pragma unroll
            for (int r = 0; r < 8; ++r)
                cm[r] = fmaxf(cm[r], __shfl_xor(cm[r], mask, 32));
        v8f alpha, p0, p1, rs;
        #pragma unroll
        for (int r = 0; r < 8; ++r) {
            const float nm = fmaxf(m_run[r], cm[r]);
            alpha[r] = __expf(m_run[r] - nm);
            m_run[r] = nm;
            p0[r] = __expf(st0[r] - nm);
            p1[r] = __expf(st1[r] - nm);
            rs[r] = p0[r] + p1[r];
        }
        #pragma unroll
        for (int mask = 1; mask <= 8; mask <<= 1)
            #pragma unroll
            for (int r = 0; r < 8; ++r)
                rs[r] += __shfl_xor(rs[r], mask, 32);
        #pragma unroll
        for (int r = 0; r < 8; ++r) {
            l_run[r] = l_run[r] * alpha[r] + rs[r];
            o0[r] *= alpha[r]; o1[r] *= alpha[r];
            o2[r] *= alpha[r]; o3[r] *= alpha[r];
        }

        // ---- re-layout P: C-layout -> LDS -> A-operand layout ----
        #pragma unroll
        for (int r = 0; r < 8; ++r) {
            Plds[wid][roff + r][colN]      = (_Float16)p0[r];
            Plds[wid][roff + r][16 + colN] = (_Float16)p1[r];
        }
        v16h ap;
        {
            const _Float16* pr = &Plds[wid][colN][0];
            v8h lo = *(const v8h*)(pr + kbA);
            v8h hi = *(const v8h*)(pr + kbA + 16);
            #pragma unroll
            for (int e = 0; e < 8; ++e) { ap[e] = lo[e]; ap[8 + e] = hi[e]; }
        }

        // ---- O += P @ V   (V^T layout makes B-operand loads contiguous) ----
        #pragma unroll
        for (int t2 = 0; t2 < 4; ++t2) {
            const _Float16* vrow = vbase + (size_t)(t2 * 16 + colN) * Lc + kk + kbB;
            v16h bm;
            v8h lo = *(const v8h*)(vrow);
            v8h hi = *(const v8h*)(vrow + 16);
            #pragma unroll
            for (int e = 0; e < 8; ++e) { bm[e] = lo[e]; bm[8 + e] = hi[e]; }
            v8f& oo = (t2 == 0) ? o0 : (t2 == 1) ? o1 : (t2 == 2) ? o2 : o3;
            oo = __builtin_amdgcn_wmma_f32_16x16x32_f16(
                false, ap, false, bm, (short)0, oo, false, false);
        }
    }

    // ---- normalize, store attn output as f16 in [B, L, D] ----
    v8f inv;
    #pragma unroll
    for (int r = 0; r < 8; ++r) inv[r] = 1.0f / l_run[r];
    #pragma unroll
    for (int t2 = 0; t2 < 4; ++t2) {
        v8f oo = (t2 == 0) ? o0 : (t2 == 1) ? o1 : (t2 == 2) ? o2 : o3;
        #pragma unroll
        for (int r = 0; r < 8; ++r)
            obuf[((size_t)(b * Lc + q0 + roff + r)) * Dc + h * HDc + t2 * 16 + colN]
                = (_Float16)(oo[r] * inv[r]);
    }
}

// ---------------------------------------------------------------------------
// Kernel 3: output projection.  out = attn_out(f16) @ Wo^T + bo (f32 out).
// One wave computes a 32(M) x 64(N) tile, all-f16 operands.
// ---------------------------------------------------------------------------
__global__ __launch_bounds__(256) void out_proj_kernel(
    const _Float16* __restrict__ obuf, const _Float16* __restrict__ Woh,
    const float* __restrict__ bo, float* __restrict__ out)
{
    const int lane = threadIdx.x & 31;
    const int wid  = threadIdx.x >> 5;
    const int wg   = blockIdx.x * 8 + wid;       // 0 .. 2047
    const int m0   = (wg >> 4) * 32;
    const int n0   = (wg & 15) * 64;

    const int colN = lane & 15;
    const int half = lane >> 4;
    const int roff = half * 8;
    const int kbA  = half * 8;
    const int kbB  = half * 8;

    const _Float16* ar0 = obuf + (size_t)(m0 + colN) * Dc;
    const _Float16* ar1 = obuf + (size_t)(m0 + 16 + colN) * Dc;

    v8f acc[8];
    #pragma unroll
    for (int i = 0; i < 8; ++i) acc[i] = {};

    for (int k0 = 0; k0 < Dc; k0 += 32) {
        v16h a0, a1;
        {
            v8h lo = *(const v8h*)(ar0 + k0 + kbA);
            v8h hi = *(const v8h*)(ar0 + k0 + kbA + 16);
            #pragma unroll
            for (int e = 0; e < 8; ++e) { a0[e] = lo[e]; a0[8 + e] = hi[e]; }
            lo = *(const v8h*)(ar1 + k0 + kbA);
            hi = *(const v8h*)(ar1 + k0 + kbA + 16);
            #pragma unroll
            for (int e = 0; e < 8; ++e) { a1[e] = lo[e]; a1[8 + e] = hi[e]; }
        }
        #pragma unroll
        for (int t = 0; t < 4; ++t) {
            const _Float16* wr = Woh + (size_t)(n0 + t * 16 + colN) * Dc + k0 + kbB;
            v16h bm;
            v8h lo = *(const v8h*)(wr);
            v8h hi = *(const v8h*)(wr + 16);
            #pragma unroll
            for (int e = 0; e < 8; ++e) { bm[e] = lo[e]; bm[8 + e] = hi[e]; }
            acc[t] = __builtin_amdgcn_wmma_f32_16x16x32_f16(
                false, a0, false, bm, (short)0, acc[t], false, false);
            acc[4 + t] = __builtin_amdgcn_wmma_f32_16x16x32_f16(
                false, a1, false, bm, (short)0, acc[4 + t], false, false);
        }
    }

    #pragma unroll
    for (int mi = 0; mi < 2; ++mi) {
        #pragma unroll
        for (int t = 0; t < 4; ++t) {
            const v8f a = acc[mi * 4 + t];
            const int n = n0 + t * 16 + colN;
            const float bv_ = bo[n];
            #pragma unroll
            for (int r = 0; r < 8; ++r)
                out[(size_t)(m0 + mi * 16 + roff + r) * Dc + n] = a[r] + bv_;
        }
    }
}

// ---------------------------------------------------------------------------
extern "C" void kernel_launch(void* const* d_in, const int* in_sizes, int n_in,
                              void* d_out, int out_size, void* d_ws, size_t ws_size,
                              hipStream_t stream)
{
    const float* x   = (const float*)d_in[0];
    const float* rel = (const float*)d_in[1];
    const float* Wq  = (const float*)d_in[2];
    const float* bq  = (const float*)d_in[3];
    const float* Wk  = (const float*)d_in[4];
    const float* bk  = (const float*)d_in[5];
    const float* Wv  = (const float*)d_in[6];
    const float* bv  = (const float*)d_in[7];
    const float* Wo  = (const float*)d_in[8];
    const float* bo  = (const float*)d_in[9];
    float* out = (float*)d_out;

    char* ws = (char*)d_ws;
    const size_t segBLD = (size_t)Bc * Hc * Lc * HDc;   // 4 Mi elements
    const size_t segW   = (size_t)Dc * Dc;              // 1 Mi elements
    size_t off = 0;
    _Float16* xh   = (_Float16*)(ws + off); off += segBLD * 2;
    _Float16* Wqh  = (_Float16*)(ws + off); off += segW * 2;
    _Float16* Wkh  = (_Float16*)(ws + off); off += segW * 2;
    _Float16* Wvh  = (_Float16*)(ws + off); off += segW * 2;
    _Float16* Woh  = (_Float16*)(ws + off); off += segW * 2;
    _Float16* qbuf = (_Float16*)(ws + off); off += segBLD * 2;
    _Float16* kbuf = (_Float16*)(ws + off); off += segBLD * 2;
    _Float16* vT   = (_Float16*)(ws + off); off += segBLD * 2;
    _Float16* obuf = (_Float16*)(ws + off); off += segBLD * 2;

    // one-shot f32 -> f16 conversions (all L2-resident afterwards)
    cvt_f16_kernel<<<(int)(segBLD / 4 / 256), 256, 0, stream>>>(x,  xh,  (int)(segBLD / 4));
    cvt_f16_kernel<<<(int)(segW   / 4 / 256), 256, 0, stream>>>(Wq, Wqh, (int)(segW / 4));
    cvt_f16_kernel<<<(int)(segW   / 4 / 256), 256, 0, stream>>>(Wk, Wkh, (int)(segW / 4));
    cvt_f16_kernel<<<(int)(segW   / 4 / 256), 256, 0, stream>>>(Wv, Wvh, (int)(segW / 4));
    cvt_f16_kernel<<<(int)(segW   / 4 / 256), 256, 0, stream>>>(Wo, Woh, (int)(segW / 4));

    // 3 projections * 128 M32-tiles * 16 N64-tiles = 6144 waves / 8 per block
    qkv_proj_kernel<<<768, 256, 0, stream>>>(xh, Wqh, bq, Wkh, bk, Wvh, bv,
                                             qbuf, kbuf, vT);
    // 32 (b,h) * 128 q-tiles = 4096 waves / 8 per block
    attn_kernel<<<512, 256, 0, stream>>>(qbuf, kbuf, vT, rel, obuf);
    // 128 M32-tiles * 16 N64-tiles = 2048 waves / 8 per block
    out_proj_kernel<<<256, 256, 0, stream>>>(obuf, Woh, bo, out);
}
